// PAE_77214922048026
// MI455X (gfx1250) — compile-verified, matrix operand order checked by reading, not run on hardware
//
#include <hip/hip_runtime.h>
#include <hip/hip_bf16.h>
#include <math.h>

typedef __bf16 bf16_t;
typedef __attribute__((ext_vector_type(16))) __bf16 v16bf;
typedef __attribute__((ext_vector_type(8)))  float  v8f;

// ---------------- problem constants ----------------
#define NB      16      // batch
#define NCH     72      // in channels
#define FULL    240     // sequence length
#define TT      121     // window length
#define PADW    60      // (T-1)/2
#define LP      360     // FULL + 2*PAD
#define NC      240     // curves per batch
#define NWIN    (NB*NC) // 3840 windows
#define INTER   24
#define EMB     8
#define TPI_F   6.28318530717958647692f

// ---------------- workspace layout (bytes) ----------------
#define OFF_XPB   0u                                   // bf16 [16][72][360]      829,440
#define OFF_W1B   829440u                              // bf16 [32][72][128]      589,824
#define OFF_W2B   1419264u                             // bf16 [16][24][128]       98,304
#define OFF_LAT   1517568u                             // f32  [3840][8][121]  14,868,480
#define OFF_P     16386048u                            // f32  [3840*8]           122,880
#define OFF_F     16508928u
#define OFF_A     16631808u
#define OFF_B     16754688u
#define OFF_SIG   16877568u                            // f32  [16][8][360]       184,320
#define OFF_DDEC  17061888u                            // f32  [16][24][240]      368,640
#define OFF_DT    17430528u                            // f32  [16][24][240]      368,640
#define OFF_BN    17799168u                            // f32  [48]
// total ~17.8 MB

// ---------------- CDNA5 async global->LDS helpers ----------------
// GLOBAL_LOAD_ASYNC_TO_LDS_B128: per-lane, LDS dest address from VGPR (byte
// offset into workgroup LDS alloc), global source from 64-bit VGPR pair.
// INST_OFFSET is added to BOTH addresses, so one base + offset:16 covers 32B.
__device__ __forceinline__ void async_lds_copy32(unsigned lds_off, const void* gsrc) {
    unsigned long long ga = (unsigned long long)(uintptr_t)gsrc;
    asm volatile("global_load_async_to_lds_b128 %0, %1, off"
                 :: "v"(lds_off), "v"(ga) : "memory");
    asm volatile("global_load_async_to_lds_b128 %0, %1, off offset:16"
                 :: "v"(lds_off), "v"(ga) : "memory");
}
__device__ __forceinline__ void wait_async0() {
    asm volatile("s_wait_asynccnt 0" ::: "memory");
}
__device__ __forceinline__ unsigned lds_offset_of(const void* p) {
    // generic pointers to __shared__ carry the LDS byte offset in their low 32 bits
    return (unsigned)(uintptr_t)p;
}

// ---------------- WMMA fragment helpers ----------------
// 16-bit A matrix 16x32 (MxK): lanes 0-15 -> M=lane, K in {0..7,16..23};
// lanes 16-31 -> M=lane-16, K in {8..15,24..31}. Element e -> K = kb + (e<8?e:e+8).
__device__ __forceinline__ v16bf load_frag_16x32(const bf16_t* base, int row_stride, int lane) {
    int m  = lane & 15;
    int kb = (lane >> 4) ? 8 : 0;
    const bf16_t* p = base + m * row_stride;
    v16bf r;
#pragma unroll
    for (int e = 0; e < 16; ++e) r[e] = p[kb + (e < 8 ? e : e + 8)];
    return r;
}
// B matrix 32x16 (KxN): symmetric layout; our LDS layout makes K and N unit-stride.
__device__ __forceinline__ v16bf load_frag_B(const bf16_t* rowbase, int lane) {
    int n  = lane & 15;
    int kb = (lane >> 4) ? 8 : 0;
    const bf16_t* p = rowbase + n;
    v16bf r;
#pragma unroll
    for (int e = 0; e < 16; ++e) r[e] = p[kb + (e < 8 ? e : e + 8)];
    return r;
}

// ---------------- prep: pad/convert inputs ----------------
__global__ __launch_bounds__(256) void pae_prep(const float* __restrict__ x,
                                                const float* __restrict__ w1,
                                                const float* __restrict__ w2,
                                                bf16_t* __restrict__ xpb,
                                                bf16_t* __restrict__ w1b,
                                                bf16_t* __restrict__ w2b,
                                                float* __restrict__ bn_acc) {
    const int NX  = NB * NCH * LP;   // 414720
    const int NW1 = 32 * NCH * 128;  // 294912
    const int NW2 = 16 * INTER * 128;// 49152
    int g = blockIdx.x * 256 + threadIdx.x;
    if (g < NX) {
        int s = g % LP; int r = g / LP; int i = r % NCH; int b = r / NCH;
        float v;
        if (s < PADW)             v = x[(b * NCH + i) * FULL + (FULL - PADW) + s];
        else if (s < PADW + FULL) v = x[(b * NCH + i) * FULL + s - PADW];
        else                      v = x[(b * NCH + i) * FULL + s - (PADW + FULL)];
        xpb[g] = (bf16_t)v;
    } else if (g < NX + NW1) {
        int q = g - NX; int k = q % 128; int r = q / 128; int i = r % NCH; int o = r / NCH;
        float v = (o < INTER && k < TT) ? w1[(o * NCH + i) * TT + k] : 0.0f;
        w1b[q] = (bf16_t)v;
    } else if (g < NX + NW1 + NW2) {
        int q = g - NX - NW1; int k = q % 128; int r = q / 128; int i = r % INTER; int e = r / INTER;
        float v = (e < EMB && k < TT) ? w2[(e * INTER + i) * TT + k] : 0.0f;
        w2b[q] = (bf16_t)v;
    } else if (g < NX + NW1 + NW2 + 48) {
        bn_acc[g - (NX + NW1 + NW2)] = 0.0f;
    }
}

// ---------------- fused encoder: conv1 (WMMA) + LN + tanh + conv2 (WMMA) ----
// one workgroup per window, 256 threads = 8 waves.
__global__ __launch_bounds__(256) void pae_encoder(const bf16_t* __restrict__ xpb,
                                                   const bf16_t* __restrict__ w1b,
                                                   const bf16_t* __restrict__ w2b,
                                                   const float* __restrict__ enc_b1,
                                                   const float* __restrict__ ln_alpha,
                                                   const float* __restrict__ ln_beta,
                                                   const float* __restrict__ enc_b2,
                                                   float* __restrict__ latent_ws) {
    __shared__ unsigned char smem[NCH * 256 * 2];  // 36,864 B window buffer, phase-reused
    __shared__ bf16_t wsl[2][32 * 128];            // 2 x 8 KB double-buffered weight slice
    bf16_t* yw = (bf16_t*)smem;                    // [72][256], data in cols 60..180

    const int n = blockIdx.x;
    const int b = n / NC;
    const int c = n % NC;
    const int tid  = threadIdx.x;
    const int lane = tid & 31;
    const int wave = tid >> 5;

    // per-thread chunk of the 8KB weight slice: 16 bf16 (32 B)
    const int so = tid >> 3;          // o row 0..31
    const int sk = (tid & 7) * 16;    // k col 0..112

    // stage window into LDS (zero apron implements the conv's zero padding):
    // yw[i*256 + 60 + tau] = xp[b,i,c+tau]   => B index = i*256 + (t + k)
    for (int j = tid; j < NCH * 256; j += 256) yw[j] = (bf16_t)0.0f;
    __syncthreads();
    for (int j = tid; j < NCH * TT; j += 256) {
        int i = j / TT, tau = j - i * TT;
        yw[i * 256 + PADW + tau] = xpb[(b * NCH + i) * LP + c + tau];
    }
    // preload weight slice i=0 asynchronously into wsl[0]
    async_lds_copy32(lds_offset_of(&wsl[0][so * 128 + sk]),
                     w1b + (so * NCH + 0) * 128 + sk);
    wait_async0();
    __syncthreads();

    // ---- conv1: h[24][121] = W1 (24x(72*121)) x Y ----
    // wave w handles t-tile t0 = 16*w for both o-tiles (o 0..15, 16..31)
    v8f acc0 = {};
    v8f acc1 = {};
    const int t0 = wave * 16;
    for (int i = 0; i < NCH; ++i) {
        const int cur = i & 1;
        if (i + 1 < NCH) // overlap next slice transfer with this slice's WMMAs
            async_lds_copy32(lds_offset_of(&wsl[cur ^ 1][so * 128 + sk]),
                             w1b + (so * NCH + (i + 1)) * 128 + sk);
        const bf16_t* wa = &wsl[cur][0];
#pragma unroll
        for (int kc = 0; kc < 128; kc += 32) {
            v16bf bf = load_frag_B(yw + i * 256 + t0 + kc, lane);
            v16bf a0 = load_frag_16x32(wa + kc,            128, lane);
            v16bf a1 = load_frag_16x32(wa + 16 * 128 + kc, 128, lane);
            acc0 = __builtin_amdgcn_wmma_f32_16x16x32_bf16(false, a0, false, bf, (short)0, acc0, false, false);
            acc1 = __builtin_amdgcn_wmma_f32_16x16x32_bf16(false, a1, false, bf, (short)0, acc1, false, false);
        }
        wait_async0();     // next slice landed in LDS
        __syncthreads();   // and all waves are done reading the current one
    }

    float*  hsm  = (float*)smem;                   // [32][128] f32 (16,384 B)
    bf16_t* hl   = (bf16_t*)(smem + 16384);        // [24][256] bf16 (12,288 B)
    float*  mstd = (float*)(smem + 16384 + 12288); // 24 means + 24 (std+eps)

    { // store D tiles (+ bias). C/D layout: vgpr v -> row v (+8 for lanes>=16), lane&15 -> col
        int t  = t0 + (lane & 15);
        int hi = (lane >> 4) * 8;
#pragma unroll
        for (int v = 0; v < 8; ++v) {
            int o0 = v + hi;
            int o1 = 16 + v + hi;
            hsm[o0 * 128 + t] = acc0[v] + ((o0 < INTER) ? enc_b1[o0] : 0.0f);
            hsm[o1 * 128 + t] = acc1[v] + ((o1 < INTER) ? enc_b1[o1] : 0.0f);
        }
    }
    for (int j = tid; j < INTER * 256; j += 256) hl[j] = (bf16_t)0.0f;
    __syncthreads();

    // LayerNorm stats (ddof=1) over t per channel
    if (tid < INTER) {
        float s = 0.0f;
        for (int t = 0; t < TT; ++t) s += hsm[tid * 128 + t];
        float m = s / (float)TT;
        float q = 0.0f;
        for (int t = 0; t < TT; ++t) { float d = hsm[tid * 128 + t] - m; q += d * d; }
        mstd[tid]         = m;
        mstd[INTER + tid] = sqrtf(q / (float)(TT - 1)) + 1e-5f;
    }
    __syncthreads();

    for (int j = tid; j < INTER * TT; j += 256) {
        int o = j / TT, t = j - o * TT;
        float hn = (hsm[o * 128 + t] - mstd[o]) / mstd[INTER + o] * ln_alpha[t] + ln_beta[t];
        hl[o * 256 + PADW + t] = (bf16_t)tanhf(hn);
    }
    __syncthreads();

    // ---- conv2: latent[8][121] = W2 (8x(24*121)) x Hln ----
    v8f accL = {};
    for (int i = 0; i < INTER; ++i) {
#pragma unroll
        for (int kc = 0; kc < 128; kc += 32) {
            v16bf bf = load_frag_B(hl + i * 256 + t0 + kc, lane);
            v16bf af = load_frag_16x32(w2b + i * 128 + kc, INTER * 128, lane);
            accL = __builtin_amdgcn_wmma_f32_16x16x32_bf16(false, af, false, bf, (short)0, accL, false, false);
        }
    }
    {
        int t = t0 + (lane & 15);
        if (lane < 16 && t < TT) { // rows 0..7 live in lanes 0..15
#pragma unroll
            for (int v = 0; v < 8; ++v)
                latent_ws[(n * EMB + v) * TT + t] = accL[v] + enc_b2[v];
        }
    }
}

// ---------------- spectral params: DFT + fc/atan2, one wave per (n,e) -------
__global__ __launch_bounds__(256) void pae_spectral(const float* __restrict__ latent_ws,
                                                    const float* __restrict__ fc_w,
                                                    const float* __restrict__ fc_b,
                                                    float* __restrict__ p_ws, float* __restrict__ f_ws,
                                                    float* __restrict__ a_ws, float* __restrict__ b_ws) {
    __shared__ float lat[EMB * TT];
    const int n = blockIdx.x, tid = threadIdx.x, lane = tid & 31, e = tid >> 5;
    for (int j = tid; j < EMB * TT; j += 256) lat[j] = latent_ws[n * (EMB * TT) + j];
    __syncthreads();
    const float* row = lat + e * TT;

    float sb = 0.0f, v0 = 0.0f, v1 = 0.0f;
    for (int t = lane; t < TT; t += 32) {
        float xv = row[t];
        sb += xv;
        v0 += xv * fc_w[(e * 2 + 0) * TT + t];
        v1 += xv * fc_w[(e * 2 + 1) * TT + t];
    }
    float sp = 0.0f, spf = 0.0f;
#pragma unroll
    for (int pass = 0; pass < 2; ++pass) {
        int jj = lane + pass * 32;
        if (jj >= 1 && jj <= 60) {
            float w = TPI_F * (float)jj / (float)TT;
            float re = 0.0f, im = 0.0f;
            for (int t = 0; t < TT; ++t) {
                float s, cs;
                __sincosf(w * (float)t, &s, &cs);
                re += row[t] * cs;
                im -= row[t] * s;
            }
            float pw = re * re + im * im;
            sp  += pw;
            spf += pw * (0.5f * (float)jj); // FREQS[j-1] == j/2
        }
    }
#pragma unroll
    for (int o = 16; o > 0; o >>= 1) {
        sb  += __shfl_xor(sb,  o, 32);
        v0  += __shfl_xor(v0,  o, 32);
        v1  += __shfl_xor(v1,  o, 32);
        sp  += __shfl_xor(sp,  o, 32);
        spf += __shfl_xor(spf, o, 32);
    }
    if (lane == 0) {
        int idx = n * EMB + e;
        f_ws[idx] = spf / sp;
        a_ws[idx] = 2.0f * sqrtf(sp) / (float)TT;
        b_ws[idx] = sb / (float)TT;
        p_ws[idx] = atan2f(v1 + fc_b[e * 2 + 1], v0 + fc_b[e * 2 + 0]) / TPI_F;
    }
}

// ---------------- sinusoid resynthesis + overlap-add ----------------------
__global__ __launch_bounds__(256) void pae_osc(const float* __restrict__ p_ws, const float* __restrict__ f_ws,
                                               const float* __restrict__ a_ws, const float* __restrict__ b_ws,
                                               float* __restrict__ signal_ws) {
    __shared__ float acc[LP];
    const int bi = blockIdx.x;          // 0..127
    const int b = bi >> 3, e = bi & 7;
    const int tid = threadIdx.x;
    for (int j = tid; j < LP; j += 256) acc[j] = 0.0f;
    __syncthreads();
    if (tid < NC) {
        const int idx = (b * NC + tid) * EMB + e;
        const float P = p_ws[idx], F = f_ws[idx], A = a_ws[idx], Bb = b_ws[idx];
        for (int t = 0; t < TT; ++t) {
            float arg = -1.0f + (float)t * (1.0f / 60.0f); // linspace(-1,1,121)
            float val = A * __sinf(TPI_F * (F * arg + P)) + Bb;
            atomicAdd(&acc[tid + t], val);
        }
    }
    __syncthreads();
    for (int s = tid; s < LP; s += 256) {
        float w = (s < TT) ? (float)(s + 1) : ((s > LP - TT) ? (float)(LP - s) : (float)TT);
        signal_ws[(b * EMB + e) * LP + s] = acc[s] / w;
    }
}

// ---------------- decoder conv1 + BN statistics ---------------------------
__global__ __launch_bounds__(256) void pae_dconv1(const float* __restrict__ signal_ws,
                                                  const float* __restrict__ dec_w1,
                                                  const float* __restrict__ dec_b1,
                                                  float* __restrict__ d_dec,
                                                  float* __restrict__ bn_acc) {
    int g = blockIdx.x * 256 + threadIdx.x; // 16*24*240 = 92160
    if (g >= NB * INTER * FULL) return;
    int t = g % FULL; int r = g / FULL; int oc = r % INTER; int b = r / INTER;
    float acc = dec_b1[oc];
    for (int ic = 0; ic < EMB; ++ic) {
        const float* sc = signal_ws + (b * EMB + ic) * LP + PADW; // yc row
        const float* wr = dec_w1 + (oc * EMB + ic) * TT;
        for (int k = 0; k < TT; ++k) {
            int tau = t + k - PADW;
            if (tau >= 0 && tau < FULL) acc += sc[tau] * wr[k];
        }
    }
    d_dec[g] = acc;
    atomicAdd(&bn_acc[oc], acc);
    atomicAdd(&bn_acc[INTER + oc], acc * acc);
}

// ---------------- BN apply + tanh ----------------------------------------
__global__ __launch_bounds__(256) void pae_bnapply(const float* __restrict__ d_dec,
                                                   const float* __restrict__ bn_acc,
                                                   const float* __restrict__ gamma,
                                                   const float* __restrict__ beta,
                                                   float* __restrict__ dt_dec) {
    int g = blockIdx.x * 256 + threadIdx.x;
    if (g >= NB * INTER * FULL) return;
    int oc = (g / FULL) % INTER;
    const float inv_n = 1.0f / (float)(NB * FULL);
    float m = bn_acc[oc] * inv_n;
    float v = bn_acc[INTER + oc] * inv_n - m * m;
    float d = (d_dec[g] - m) * rsqrtf(v + 1e-5f) * gamma[oc] + beta[oc];
    dt_dec[g] = tanhf(d);
}

// ---------------- decoder conv2 -> y_out (d_out region 0) ----------------
__global__ __launch_bounds__(256) void pae_dconv2(const float* __restrict__ dt_dec,
                                                  const float* __restrict__ dec_w2,
                                                  const float* __restrict__ dec_b2,
                                                  float* __restrict__ out) {
    int g = blockIdx.x * 256 + threadIdx.x; // 16*72*240 = 276480
    if (g >= NB * NCH * FULL) return;
    int t = g % FULL; int r = g / FULL; int o = r % NCH; int b = r / NCH;
    float acc = dec_b2[o];
    for (int ic = 0; ic < INTER; ++ic) {
        const float* dr = dt_dec + (b * INTER + ic) * FULL;
        const float* wr = dec_w2 + (o * INTER + ic) * TT;
        for (int k = 0; k < TT; ++k) {
            int tau = t + k - PADW;
            if (tau >= 0 && tau < FULL) acc += dr[tau] * wr[k];
        }
    }
    out[g] = acc;
}

// ---------------- gather remaining outputs --------------------------------
__global__ __launch_bounds__(256) void pae_gather(const float* __restrict__ latent_ws,
                                                  const float* __restrict__ signal_ws,
                                                  const float* __restrict__ p_ws, const float* __restrict__ f_ws,
                                                  const float* __restrict__ a_ws, const float* __restrict__ b_ws,
                                                  float* __restrict__ out) {
    const int O_YOUT = 0;
    const int O_LAT  = NB * NCH * FULL;          // 276480
    const int N_LAT  = NB * EMB * TT;            // 15488
    const int O_SIG  = O_LAT + N_LAT;            // 291968
    const int N_SIG  = NB * EMB * TT;            // 15488
    const int O_PAR  = O_SIG + N_SIG;            // 307456
    const int N_PAR  = NB * EMB;                 // 128 each, x4
    const int O_YP   = O_PAR + 4 * N_PAR;        // 307968
    const int N_YP   = NB * NCH * TT;            // 139392

    int g = blockIdx.x * 256 + threadIdx.x;
    if (g < N_LAT) {
        int t = g % TT; int r = g / TT; int e = r % EMB; int b = r / EMB;
        out[O_LAT + g] = latent_ws[((b * NC) * EMB + e) * TT + t];
    } else if (g < N_LAT + N_SIG) {
        int q = g - N_LAT;
        int t = q % TT; int r = q / TT; int e = r % EMB; int b = r / EMB;
        out[O_SIG + q] = signal_ws[(b * EMB + e) * LP + t];
    } else if (g < N_LAT + N_SIG + 4 * N_PAR) {
        int q = g - N_LAT - N_SIG;
        int which = q / N_PAR; int idx = q % N_PAR;
        int e = idx % EMB; int b = idx / EMB;
        const float* src = (which == 0) ? p_ws : (which == 1) ? f_ws : (which == 2) ? a_ws : b_ws;
        out[O_PAR + q] = src[(b * NC) * EMB + e];
    } else if (g < N_LAT + N_SIG + 4 * N_PAR + N_YP) {
        int q = g - (N_LAT + N_SIG + 4 * N_PAR);
        int t = q % TT; int r = q / TT; int o = r % NCH; int b = r / NCH;
        out[O_YP + q] = out[O_YOUT + (b * NCH + o) * FULL + t];
    }
}

// ---------------- host launcher -------------------------------------------
extern "C" void kernel_launch(void* const* d_in, const int* in_sizes, int n_in,
                              void* d_out, int out_size, void* d_ws, size_t ws_size,
                              hipStream_t stream) {
    (void)in_sizes; (void)n_in; (void)out_size; (void)ws_size;
    const float* x      = (const float*)d_in[0];
    const float* enc_w1 = (const float*)d_in[1];
    const float* enc_b1 = (const float*)d_in[2];
    const float* ln_a   = (const float*)d_in[3];
    const float* ln_b   = (const float*)d_in[4];
    const float* enc_w2 = (const float*)d_in[5];
    const float* enc_b2 = (const float*)d_in[6];
    const float* fc_w   = (const float*)d_in[7];
    const float* fc_b   = (const float*)d_in[8];
    const float* dec_w1 = (const float*)d_in[9];
    const float* dec_b1 = (const float*)d_in[10];
    const float* bn_g   = (const float*)d_in[11];
    const float* bn_bt  = (const float*)d_in[12];
    const float* dec_w2 = (const float*)d_in[13];
    const float* dec_b2 = (const float*)d_in[14];

    char* ws = (char*)d_ws;
    bf16_t* xpb  = (bf16_t*)(ws + OFF_XPB);
    bf16_t* w1b  = (bf16_t*)(ws + OFF_W1B);
    bf16_t* w2b  = (bf16_t*)(ws + OFF_W2B);
    float*  lat  = (float*)(ws + OFF_LAT);
    float*  pws  = (float*)(ws + OFF_P);
    float*  fws  = (float*)(ws + OFF_F);
    float*  aws  = (float*)(ws + OFF_A);
    float*  bws  = (float*)(ws + OFF_B);
    float*  sig  = (float*)(ws + OFF_SIG);
    float*  ddec = (float*)(ws + OFF_DDEC);
    float*  dtd  = (float*)(ws + OFF_DT);
    float*  bn   = (float*)(ws + OFF_BN);
    float*  out  = (float*)d_out;

    const int prep_elems = NB * NCH * LP + 32 * NCH * 128 + 16 * INTER * 128 + 48;
    pae_prep<<<(prep_elems + 255) / 256, 256, 0, stream>>>(x, enc_w1, enc_w2, xpb, w1b, w2b, bn);
    pae_encoder<<<NWIN, 256, 0, stream>>>(xpb, w1b, w2b, enc_b1, ln_a, ln_b, enc_b2, lat);
    pae_spectral<<<NWIN, 256, 0, stream>>>(lat, fc_w, fc_b, pws, fws, aws, bws);
    pae_osc<<<NB * EMB, 256, 0, stream>>>(pws, fws, aws, bws, sig);
    pae_dconv1<<<(NB * INTER * FULL + 255) / 256, 256, 0, stream>>>(sig, dec_w1, dec_b1, ddec, bn);
    pae_bnapply<<<(NB * INTER * FULL + 255) / 256, 256, 0, stream>>>(ddec, bn, bn_g, bn_bt, dtd);
    pae_dconv2<<<(NB * NCH * FULL + 255) / 256, 256, 0, stream>>>(dtd, dec_w2, dec_b2, out);
    const int gather_elems = NB * EMB * TT * 2 + 4 * NB * EMB + NB * NCH * TT;
    pae_gather<<<(gather_elems + 255) / 256, 256, 0, stream>>>(lat, sig, pws, fws, aws, bws, out);
}
